// Muti_scale_extractor_89970974916763
// MI455X (gfx1250) — matrix operand order repaired and checked by reading, not compiled, not run on hardware
//
#include <hip/hip_runtime.h>
#include <cstdint>
#include <cmath>

typedef __attribute__((ext_vector_type(16))) _Float16 v16h;
typedef __attribute__((ext_vector_type(8)))  _Float16 v8h;
typedef __attribute__((ext_vector_type(8)))  float    v8f;

#define NBATCH 2

__device__ __forceinline__ float act_apply(float v, int m) {
  if (m == 1) return v > 0.f ? v : 0.1f * v;   // leaky relu (0.1)
  if (m == 2) return v > 0.f ? v : 0.f;        // relu
  return v;
}

// ---------------------------------------------------------------- utilities
__global__ void zero_u32_kernel(uint32_t* __restrict__ p, long n) {
  long i = (long)blockIdx.x * blockDim.x + threadIdx.x;
  if (i < n) p[i] = 0u;
}

// f32 NCHW -> f16 channel-last padded [N][H+2p][W+2p][Cpad] (halo pre-zeroed)
__global__ void pack_act_kernel(const float* __restrict__ in, _Float16* __restrict__ out,
                                int C, int H, int W, int pad, int Cpad, int cofs, int actm,
                                long total) {
  long i = (long)blockIdx.x * blockDim.x + threadIdx.x;
  if (i >= total) return;
  int x = (int)(i % W); long t = i / W;
  int y = (int)(t % H); t /= H;
  int c = (int)(t % C); int n = (int)(t / C);
  int Wp = W + 2 * pad, Hp = H + 2 * pad;
  float v = act_apply(in[i], actm);
  out[((long)n * Hp * Wp + (long)(y + pad) * Wp + (x + pad)) * Cpad + cofs + c] = (_Float16)v;
}

// f32 [Cout][Cin][kk][kk] -> f16 [chunk][CoutPad][32], chunk=(kh*kk+kw)*nCb+cb
__global__ void repack_w_kernel(const float* __restrict__ w, _Float16* __restrict__ o,
                                int Cout, int CoutPad, int Cin, int nCb, int kk, long total) {
  long i = (long)blockIdx.x * blockDim.x + threadIdx.x;
  if (i >= total) return;
  int cl = (int)(i & 31);
  int oc = (int)((i >> 5) % CoutPad);
  int chunk = (int)(i / (32L * CoutPad));
  int cb = chunk % nCb, t = chunk / nCb;
  int ci = cb * 32 + cl;
  float v = 0.f;
  if (oc < Cout && ci < Cin)
    v = w[((long)(oc * Cin + ci) * kk + (t / kk)) * kk + (t % kk)];
  o[i] = (_Float16)v;
}

// -------------------------------------------------- implicit-GEMM WMMA conv
// One wave computes a 32-pixel x 16-cout tile: one A (weight) fragment feeds
// two WMMAs (two 16-pixel B fragments), fully-unrolled K loop (KK,NCB are
// compile-time).  A = weights 16x32 f16 (row = cout), B = activations 32x16
// f16 (col = pixel), D = f32 16x16.
template <int KK, int NCB>
__global__ void __launch_bounds__(32)
conv_wmma_tk(const _Float16* __restrict__ act, const _Float16* __restrict__ wt,
             const float* __restrict__ bias, const float* __restrict__ addsrc,
             float* __restrict__ out,
             int Ho, int Wo, int Wp, int stride,
             int CoutPad, int Cout, int actm, long aStride) {
  constexpr int CPAD = NCB * 32;
  int lane = threadIdx.x;
  int hi = lane >> 4;       // half-wave selector
  int lm = lane & 15;
  int tilesX = Wo >> 5;     // 32 output pixels per wave
  int tile = blockIdx.x;
  int y = tile / tilesX;
  int x0 = (tile - y * tilesX) << 5;
  int mt = blockIdx.y, nb = blockIdx.z;
  long HWo = (long)Ho * Wo;
  const _Float16* A = act + (long)nb * aStride;

  v8f c0, c1;
#pragma unroll
  for (int v = 0; v < 8; ++v) {
    int co = mt * 16 + hi * 8 + v;
    float bv = 0.f;
    if (bias) bv = bias[min(co, Cout - 1)] * (co < Cout ? 1.f : 0.f);
    c0[v] = bv; c1[v] = bv;
  }

  // weight lane base: row = cout (A-matrix), two 8-half K chunks per lane
  const _Float16* wl = wt + (long)(mt * 16 + lm) * 32 + hi * 16;
  const long wcs = (long)CoutPad * 32;  // elements per K-chunk
  int xA = x0 + lm;                     // B columns: pixels for the two tiles
  int xB = x0 + 16 + lm;

#pragma unroll
  for (int t = 0; t < KK * KK; ++t) {
    const int kh = t / KK, kw = t - kh * KK;
    const _Float16* abA =
        A + ((long)(y * stride + kh) * Wp + (xA * stride + kw)) * CPAD + hi * 16;
    const _Float16* abB =
        A + ((long)(y * stride + kh) * Wp + (xB * stride + kw)) * CPAD + hi * 16;
    __builtin_prefetch(wl + (long)(t + 1) * NCB * wcs, 0, 1);  // next tap's weights
#pragma unroll
    for (int cb = 0; cb < NCB; ++cb) {
      const _Float16* wp = wl + (long)(t * NCB + cb) * wcs;
      v8h a0 = *(const v8h*)(wp);
      v8h a1 = *(const v8h*)(wp + 8);
      v8h p0 = *(const v8h*)(abA + cb * 32);
      v8h p1 = *(const v8h*)(abA + cb * 32 + 8);
      v8h q0 = *(const v8h*)(abB + cb * 32);
      v8h q1 = *(const v8h*)(abB + cb * 32 + 8);
      v16h av, bv0, bv1;
#pragma unroll
      for (int q = 0; q < 8; ++q) {
        av[q] = a0[q];  av[q + 8] = a1[q];
        bv0[q] = p0[q]; bv0[q + 8] = p1[q];
        bv1[q] = q0[q]; bv1[q + 8] = q1[q];
      }
      c0 = __builtin_amdgcn_wmma_f32_16x16x32_f16(false, av, false, bv0,
                                                  (short)0, c0, false, false);
      c1 = __builtin_amdgcn_wmma_f32_16x16x32_f16(false, av, false, bv1,
                                                  (short)0, c1, false, false);
    }
  }

  float* O = out + (long)nb * Cout * HWo;
  const float* AD = addsrc ? addsrc + (long)nb * Cout * HWo : nullptr;
#pragma unroll
  for (int v = 0; v < 8; ++v) {
    int co = mt * 16 + hi * 8 + v;
    if (co < Cout) {
      long rb = (long)co * HWo + (long)y * Wo;
      float r0 = c0[v], r1 = c1[v];
      if (AD) { r0 += AD[rb + xA]; r1 += AD[rb + xB]; }
      O[rb + xA] = act_apply(r0, actm);
      O[rb + xB] = act_apply(r1, actm);
    }
  }
}

// -------------------------------------------------------------- DCN gather
// Builds per-pixel K=576 f16 vector: vbuf[n][y][x][tap*64 + dg*8 + cg]
// = mask * bilinear(x_group) ; feeds the K=576 WMMA GEMM.
__global__ void dcn_gather_kernel(const float* __restrict__ xf, const float* __restrict__ co,
                                  _Float16* __restrict__ vbuf, int H, int W, long total) {
  long i = (long)blockIdx.x * blockDim.x + threadIdx.x;
  if (i >= total) return;
  int xx = (int)(i % W); long t = i / W;
  int yy = (int)(t % H); t /= H;
  int tap = (int)(t % 9); t /= 9;
  int dg = (int)(t % 8); int n = (int)(t / 8);
  long HW = (long)H * W;
  const float* cb = co + (long)n * 216 * HW + (long)yy * W + xx;
  int ch = dg * 9 + tap;
  float o1 = cb[(long)(2 * ch) * HW];
  float o2 = cb[(long)(2 * ch + 1) * HW];
  float mk = cb[(long)(144 + ch) * HW];
  mk = 1.f / (1.f + expf(-mk));
  int ki = tap / 3, kj = tap - ki * 3;
  float py = (float)yy - 1.f + (float)ki + o1;
  float px = (float)xx - 1.f + (float)kj + o2;
  float y0f = floorf(py), x0f = floorf(px);
  int y0 = (int)y0f, x0 = (int)x0f;
  float wy1 = py - y0f, wx1 = px - x0f, wy0 = 1.f - wy1, wx0 = 1.f - wx1;
  int y1 = y0 + 1, x1 = x0 + 1;
  bool vy0 = (y0 >= 0 && y0 < H), vy1 = (y1 >= 0 && y1 < H);
  bool vx0 = (x0 >= 0 && x0 < W), vx1 = (x1 >= 0 && x1 < W);
  int cy0 = min(max(y0, 0), H - 1), cy1 = min(max(y1, 0), H - 1);
  int cx0 = min(max(x0, 0), W - 1), cx1 = min(max(x1, 0), W - 1);
  float w00 = (vy0 && vx0) ? wy0 * wx0 : 0.f;
  float w01 = (vy0 && vx1) ? wy0 * wx1 : 0.f;
  float w10 = (vy1 && vx0) ? wy1 * wx0 : 0.f;
  float w11 = (vy1 && vx1) ? wy1 * wx1 : 0.f;
  long i00 = (long)cy0 * W + cx0, i01 = (long)cy0 * W + cx1;
  long i10 = (long)cy1 * W + cx0, i11 = (long)cy1 * W + cx1;
  _Float16* vp = vbuf + ((long)n * HW + (long)yy * W + xx) * 576 + tap * 64 + dg * 8;
#pragma unroll
  for (int cg = 0; cg < 8; ++cg) {
    const float* xb = xf + ((long)n * 64 + dg * 8 + cg) * HW;
    float v = w00 * xb[i00] + w01 * xb[i01] + w10 * xb[i10] + w11 * xb[i11];
    vp[cg] = (_Float16)(v * mk);
  }
}

// ------------------------------------------------- pixel-adaptive depthwise
__global__ void pixelconv_kernel(const float* __restrict__ f, const float* __restrict__ k,
                                 float* __restrict__ o, int C, int H, int W, long total) {
  long i = (long)blockIdx.x * blockDim.x + threadIdx.x;
  if (i >= total) return;
  int x = (int)(i % W); long t = i / W;
  int y = (int)(t % H); t /= H;
  int c = (int)(t % C); int n = (int)(t / C);
  long HW = (long)H * W;
  const float* kb = k + (long)n * 25 * HW + (long)y * W + x;
  const float* fb = f + ((long)n * C + c) * HW;
  float acc = 0.f;
  for (int kw = 0; kw < 5; ++kw)
    for (int kh = 0; kh < 5; ++kh) {
      int yy = y + kh - 2, xx = x + kw - 2;
      if (yy >= 0 && yy < H && xx >= 0 && xx < W)
        acc += fb[(long)yy * W + xx] * kb[(long)(kw * 5 + kh) * HW];
    }
  o[i] = acc;
}

// -------------------------------------------------------- bilinear upsample
__global__ void upsample2_kernel(const float* __restrict__ in, float* __restrict__ o,
                                 int H, int W, float scale, long total) {
  long i = (long)blockIdx.x * blockDim.x + threadIdx.x;
  if (i >= total) return;
  int W2 = 2 * W, H2 = 2 * H;
  int x = (int)(i % W2); long t = i / W2;
  int y = (int)(t % H2); int c = (int)(t / H2);
  float sy = (y + 0.5f) * 0.5f - 0.5f, sx = (x + 0.5f) * 0.5f - 0.5f;
  float y0f = floorf(sy), x0f = floorf(sx);
  float fy = sy - y0f, fx = sx - x0f;
  int y0 = (int)y0f, x0 = (int)x0f;
  int y0c = min(max(y0, 0), H - 1), y1c = min(max(y0 + 1, 0), H - 1);
  int x0c = min(max(x0, 0), W - 1), x1c = min(max(x0 + 1, 0), W - 1);
  const float* ib = in + (long)c * H * W;
  float v = (1.f - fy) * ((1.f - fx) * ib[(long)y0c * W + x0c] + fx * ib[(long)y0c * W + x1c])
          +        fy  * ((1.f - fx) * ib[(long)y1c * W + x0c] + fx * ib[(long)y1c * W + x1c]);
  o[i] = v * scale;
}

__global__ void ewise_kernel(const float* __restrict__ a, const float* __restrict__ b,
                             float* __restrict__ o, int mode, long n) {
  long i = (long)blockIdx.x * blockDim.x + threadIdx.x;
  if (i >= n) return;
  float v = a[i];
  if (b) v += b[i];
  o[i] = act_apply(v, mode);
}

// =========================================================================
static inline int alignup_i(int v, int a) { return (v + a - 1) / a * a; }

extern "C" void kernel_launch(void* const* d_in, const int* in_sizes, int n_in,
                              void* d_out, int out_size, void* d_ws, size_t ws_size,
                              hipStream_t stream) {
  (void)in_sizes; (void)n_in; (void)out_size; (void)ws_size;
  struct CW { const float* w; const float* b; };
  struct DCNp { const float* b; CW off; const float* w; };
  struct DYNp { CW fk, first, fr1, fr2, ic1, ic2, rba[5], rbb[5]; };
  int cur = 0;
  auto nf = [&]() -> const float* { return (const float*)d_in[cur++]; };
  auto nCW = [&] { CW c; c.w = nf(); c.b = nf(); return c; };
  auto nDCN = [&] { DCNp d; d.b = nf(); d.off = nCW(); d.w = nf(); return d; };
  auto nDYN = [&] {
    DYNp y; y.fk = nCW(); y.first = nCW(); y.fr1 = nCW(); y.fr2 = nCW();
    y.ic1 = nCW(); y.ic2 = nCW();
    for (int i = 0; i < 5; ++i) { y.rba[i] = nCW(); y.rbb[i] = nCW(); }
    return y;
  };

  // pytree (sorted-key) flattening order
  const float* img = nf();
  CW cas1 = nCW(), cas2 = nCW();
  DCNp casd = nDCN();
  CW cfirst = nCW(), cl21 = nCW(), cl22 = nCW(), cl31 = nCW(), cl32 = nCW();
  DCNp dcn[3]; for (int i = 0; i < 3; ++i) dcn[i] = nDCN();        // l1,l2,l3
  CW fcv[2]; fcv[0] = nCW(); fcv[1] = nCW();                       // l1,l2
  DYNp ocv1[3]; for (int i = 0; i < 3; ++i) ocv1[i] = nDYN();      // l1,l2,l3
  CW ocv2[3]; for (int i = 0; i < 3; ++i) ocv2[i] = nCW();         // l1,l2,l3
  CW ocv3[2]; ocv3[0] = nCW(); ocv3[1] = nCW();                    // l1,l2

  // ------------------------------------------------------------ arena
  char* base = (char*)d_ws; size_t cursor = 0;
  auto alloc = [&](size_t bytes) -> char* {
    cursor = (cursor + 255) & ~(size_t)255;
    char* p = base + cursor; cursor += bytes; return p;
  };
  const long S1 = 128L * 128, S2 = 64L * 64, S3 = 32L * 32;
  auto f32buf = [&](long S, int C) { return (float*)alloc((size_t)NBATCH * C * S * 4); };
  float* FEATL1 = f32buf(S1, 64);
  float* FEATL2 = f32buf(S2, 64);
  float* FEATL3 = f32buf(S3, 64);
  float* OFF  = f32buf(S1, 64);
  float* FEATB = f32buf(S1, 64);
  float* UPOFF = f32buf(S1, 64);
  float* UPFEAT = f32buf(S1, 64);
  float* T1 = f32buf(S1, 64); float* T2 = f32buf(S1, 64); float* T3 = f32buf(S1, 64);
  float* T4 = f32buf(S1, 64); float* T5 = f32buf(S1, 64); float* T6 = f32buf(S1, 64);
  float* KBUF = f32buf(S1, 25);
  float* CO = f32buf(S1, 216);
  _Float16* VBUF = (_Float16*)alloc((size_t)NBATCH * S1 * 576 * 2);
  _Float16* ABUF = (_Float16*)alloc((size_t)NBATCH * 130 * 130 * 128 * 2);
  _Float16* WBUF = (_Float16*)alloc((size_t)1 << 19);

  auto gN = [&](long n) { return dim3((unsigned)((n + 255) / 256)); };
  auto zero = [&](void* p, size_t bytes) {
    long n = (long)(bytes / 4);
    zero_u32_kernel<<<gN(n), 256, 0, stream>>>((uint32_t*)p, n);
  };
  auto pack = [&](const float* src, int C, int H, int W, int pad, int Cpad, int cofs, int actm) {
    long total = (long)NBATCH * C * H * W;
    pack_act_kernel<<<gN(total), 256, 0, stream>>>(src, ABUF, C, H, W, pad, Cpad, cofs, actm, total);
  };
  auto zeroAbuf = [&](int Cpad, int H, int W, int pad) {
    zero(ABUF, (size_t)NBATCH * (H + 2 * pad) * (W + 2 * pad) * Cpad * 2);
  };
  auto repackW = [&](const float* w, int Cout, int CoutPad, int Cin, int kk) {
    int nCb = alignup_i(Cin, 32) / 32;
    long total = (long)kk * kk * nCb * CoutPad * 32;
    repack_w_kernel<<<gN(total), 256, 0, stream>>>(w, WBUF, Cout, CoutPad, Cin, nCb, kk, total);
  };
  auto runConv = [&](const _Float16* a, long aStride, int Wp, int Cpad, int kk, int stride,
                     const float* bias, int CoutPad, int Cout, int Ho, int Wo, int actm,
                     const float* addsrc, float* out) {
    dim3 g((unsigned)(Wo / 32 * Ho), (unsigned)(CoutPad / 16), NBATCH);
    int nCb = Cpad / 32;
    if (kk == 3 && nCb == 1)
      conv_wmma_tk<3, 1><<<g, 32, 0, stream>>>(a, WBUF, bias, addsrc, out, Ho, Wo, Wp,
                                               stride, CoutPad, Cout, actm, aStride);
    else if (kk == 3 && nCb == 2)
      conv_wmma_tk<3, 2><<<g, 32, 0, stream>>>(a, WBUF, bias, addsrc, out, Ho, Wo, Wp,
                                               stride, CoutPad, Cout, actm, aStride);
    else if (kk == 3 && nCb == 4)
      conv_wmma_tk<3, 4><<<g, 32, 0, stream>>>(a, WBUF, bias, addsrc, out, Ho, Wo, Wp,
                                               stride, CoutPad, Cout, actm, aStride);
    else if (kk == 1 && nCb == 18)
      conv_wmma_tk<1, 18><<<g, 32, 0, stream>>>(a, WBUF, bias, addsrc, out, Ho, Wo, Wp,
                                                stride, CoutPad, Cout, actm, aStride);
  };
  auto stdconv = [&](const float* in, int Cin, CW cw, int Cout, int H, int W, int stride,
                     int actm, const float* addsrc, float* out) {
    int Cpad = alignup_i(Cin, 32), CoutPad = alignup_i(Cout, 16);
    zeroAbuf(Cpad, H, W, 1);
    pack(in, Cin, H, W, 1, Cpad, 0, 0);
    repackW(cw.w, Cout, CoutPad, Cin, 3);
    runConv(ABUF, (long)(H + 2) * (W + 2) * Cpad, W + 2, Cpad, 3, stride,
            cw.b, CoutPad, Cout, H / stride, W / stride, actm, addsrc, out);
  };
  auto catconv = [&](const float* inA, const float* inB, CW cw, int Cout, int H, int W,
                     int actm, const float* addsrc, float* out) {
    zeroAbuf(128, H, W, 1);
    pack(inA, 64, H, W, 1, 128, 0, 0);
    pack(inB, 64, H, W, 1, 128, 64, 0);
    repackW(cw.w, Cout, alignup_i(Cout, 16), 128, 3);
    runConv(ABUF, (long)(H + 2) * (W + 2) * 128, W + 2, 128, 3, 1,
            cw.b, alignup_i(Cout, 16), Cout, H, W, actm, addsrc, out);
  };
  auto ewise = [&](const float* a, const float* b, float* o, int mode, long n) {
    ewise_kernel<<<gN(n), 256, 0, stream>>>(a, b, o, mode, n);
  };

  auto runDyn = [&](DYNp& dy, const float* x, int H, int W, float* outp) {
    long n64 = (long)NBATCH * 64 * H * W;
    stdconv(x, 64, dy.first, 64, H, W, 1, 0, nullptr, T1);          // f
    const float* r = T1;
    for (int i = 0; i < 5; ++i) {
      stdconv(r, 64, dy.rba[i], 64, H, W, 1, 2, nullptr, T3);       // relu(conv(r))
      stdconv(T3, 64, dy.rbb[i], 64, H, W, 1, 0, r, T2);            // conv + r
      r = T2;
    }
    ewise(T1, T2, T5, 0, n64);                                      // feature = f + r
    stdconv(x, 64, dy.ic1, 64, H, W, 1, 2, nullptr, T3);
    stdconv(T3, 64, dy.ic2, 64, H, W, 1, 0, nullptr, T4);           // t
    zeroAbuf(128, H, W, 1);                                         // cat(t, feature)
    pack(T4, 64, H, W, 1, 128, 0, 0);
    pack(T5, 64, H, W, 1, 128, 64, 0);
    long aStride = (long)(H + 2) * (W + 2) * 128;
    repackW(dy.fk.w, 25, 32, 128, 3);
    runConv(ABUF, aStride, W + 2, 128, 3, 1, dy.fk.b, 32, 25, H, W, 0, nullptr, KBUF);
    repackW(dy.fr1.w, 64, 64, 128, 3);
    runConv(ABUF, aStride, W + 2, 128, 3, 1, dy.fr1.b, 64, 64, H, W, 2, nullptr, T3);
    pixelconv_kernel<<<gN(n64), 256, 0, stream>>>(T4, KBUF, T6, 64, H, W, n64);
    stdconv(T3, 64, dy.fr2, 64, H, W, 1, 1, T6, outp);              // lrelu(out + res)
  };

  auto dcnPack = [&](DCNp& d, const float* x, const float* offs, int H, int W,
                     int actm, float* out) {
    stdconv(offs, 64, d.off, 216, H, W, 1, 0, nullptr, CO);
    long tg = (long)NBATCH * 8 * 9 * H * W;
    dcn_gather_kernel<<<gN(tg), 256, 0, stream>>>(x, CO, VBUF, H, W, tg);
    repackW(d.w, 64, 64, 64, 3);       // layout identical to K=576 1x1 chunk order
    runConv(VBUF, (long)H * W * 576, W, 576, 1, 1, d.b, 64, 64, H, W, actm, nullptr, out);
  };

  // ------------------------------------------------------- multi-scale feats
  stdconv(img, 3, cfirst, 64, 128, 128, 1, 1, nullptr, FEATL1);
  stdconv(FEATL1, 64, cl21, 64, 128, 128, 2, 1, nullptr, T1);
  stdconv(T1, 64, cl22, 64, 64, 64, 1, 1, nullptr, FEATL2);
  stdconv(FEATL2, 64, cl31, 64, 64, 64, 2, 1, nullptr, T1);
  stdconv(T1, 64, cl32, 64, 32, 32, 1, 1, nullptr, FEATL3);

  const float* featl[3] = { FEATL1, FEATL2, FEATL3 };
  int dims[3] = { 128, 64, 32 };
  for (int i = 3; i >= 1; --i) {
    int H = dims[i - 1], W = H;
    runDyn(ocv1[i - 1], featl[i - 1], H, W, OFF);
    if (i == 3) {
      stdconv(OFF, 64, ocv2[2], 64, H, W, 1, 1, nullptr, OFF);
    } else {
      catconv(OFF, UPOFF, ocv2[i - 1], 64, H, W, 1, nullptr, OFF);
      stdconv(OFF, 64, ocv3[i - 1], 64, H, W, 1, 1, nullptr, OFF);
    }
    if (i == 3) {
      dcnPack(dcn[2], featl[2], OFF, H, W, 1, FEATB);
    } else {
      dcnPack(dcn[i - 1], featl[i - 1], OFF, H, W, 0, T1);
      catconv(T1, UPFEAT, fcv[i - 1], 64, H, W, (i == 2) ? 1 : 0, nullptr, FEATB);
    }
    if (i > 1) {
      long tot = (long)NBATCH * 64 * (2 * H) * (2 * W);
      upsample2_kernel<<<gN(tot), 256, 0, stream>>>(OFF, UPOFF, H, W, 2.0f, tot);
      upsample2_kernel<<<gN(tot), 256, 0, stream>>>(FEATB, UPFEAT, H, W, 1.0f, tot);
    }
  }

  // --------------------------------------------------------------- cascade
  catconv(FEATB, FEATL1, cas1, 64, 128, 128, 1, nullptr, T1);
  stdconv(T1, 64, cas2, 64, 128, 128, 1, 1, nullptr, OFF);
  dcnPack(casd, FEATB, OFF, 128, 128, 1, (float*)d_out);
}
